// selfAttentionMultiHead_52012053954871
// MI455X (gfx1250) — matrix operand order repaired and checked by reading, not compiled, not run on hardware
//
#include <hip/hip_runtime.h>
#include <hip/hip_bf16.h>
#include <stdint.h>

// Fused QKV-projection + per-position cross-head attention.
// B=16, S=4096, D=1024, H=16, dh=64. M = B*S = 65536 rows.
//
// kernel 0 (if ws available): Wq|Wk|Wv (fp32, [k][n]) -> wt (bf16, [mat][n][k]) in d_ws.
// kernel 1: per workgroup (256 thr / 8 waves) owns MTILE=32 rows:
//   phase 1: (32 x 1024) @ (1024 x 3072) on v_wmma_f32_16x16x32_bf16,
//            weights staged into double-buffered LDS via global_load_async_to_lds_b128
//            (ASYNCcnt-tracked, overlapped with WMMA), QKV kept bf16 in LDS.
//   phase 2: 16x16 scores (2 WMMA) -> lane-shuffle softmax -> 16x64 out (4 WMMA).

#define BDIM  256
#define MTILE 32
#define DK    1024
#define N3    3072
#define KC    128

typedef __attribute__((ext_vector_type(16))) __bf16 v16bf;
typedef __attribute__((ext_vector_type(8)))  __bf16 v8bf;
typedef __attribute__((ext_vector_type(8)))  float  v8f;
typedef __attribute__((ext_vector_type(4)))  float  v4f;

union V16 { v16bf v; v8bf h[2]; };

__device__ __forceinline__ v8bf lds_ld8(const __bf16* p) {
    return *(const v8bf*)p;
}
__device__ __forceinline__ v16bf lds_ld16(const __bf16* p) {
    return *(const v16bf*)p;
}

// ---------------- kernel 0: weight convert + transpose (fp32 [k][n] -> bf16 [n][k]) ----
__global__ __launch_bounds__(256, 2)
void weight_tr_cvt_kernel(const float* __restrict__ Wq,
                          const float* __restrict__ Wk,
                          const float* __restrict__ Wv,
                          __bf16* __restrict__ wt)
{
    __shared__ float tile[64][65];              // padded vs bank conflicts
    const int bid  = blockIdx.x;                // 0..767 : 3 mats x 16x16 tiles
    const int mat  = bid >> 8;
    const int tidx = bid & 255;
    const int tk   = (tidx >> 4) * 64;          // k tile base
    const int tn   = (tidx & 15) * 64;          // n tile base
    const float* W = (mat == 0) ? Wq : ((mat == 1) ? Wk : Wv);
    const int t = threadIdx.x;

    #pragma unroll
    for (int j = 0; j < 4; ++j) {               // load 64x64 fp32, coalesced in n
        const int kl = j * 16 + (t >> 4);
        const int nl = (t & 15) * 4;
        v4f d = *(const v4f*)(W + (size_t)(tk + kl) * 1024 + tn + nl);
        tile[kl][nl + 0] = d.x;
        tile[kl][nl + 1] = d.y;
        tile[kl][nl + 2] = d.z;
        tile[kl][nl + 3] = d.w;
    }
    __syncthreads();

    __bf16* wbase = wt + (size_t)mat * 1024 * 1024;
    #pragma unroll
    for (int j = 0; j < 8; ++j) {               // write bf16 [n][k], packed pairs
        const int nl = t >> 2;                  // 0..63
        const int pp = (t & 3) * 8 + j;         // 0..31 uint pairs per row
        const int k0 = pp * 2;
        union { __bf16 h[2]; uint32_t u; } pk;
        pk.h[0] = (__bf16)tile[k0 + 0][nl];
        pk.h[1] = (__bf16)tile[k0 + 1][nl];
        *(uint32_t*)(wbase + (size_t)(tn + nl) * 1024 + tk + k0) = pk.u;
    }
}

// ---------------- kernel 1: fused GEMM + attention ----------------
template<bool USE_WT>
__global__ __launch_bounds__(BDIM, 1)
void fused_mha_kernel(const float* __restrict__ x,
                      const float* __restrict__ Wq, const float* __restrict__ bq,
                      const float* __restrict__ Wk, const float* __restrict__ bk,
                      const float* __restrict__ Wv, const float* __restrict__ bv,
                      const __bf16* __restrict__ wt,
                      float* __restrict__ out)
{
    // 64 KB + 192 KB + 32 KB = 288 KB (<= 320 KB CDNA5 workgroup LDS)
    __shared__ __align__(16) __bf16 sA[MTILE * DK];     // x tile, bf16
    __shared__ __align__(16) __bf16 sQKV[MTILE * N3];   // [row][ q | k | v ]
    __shared__ __align__(16) __bf16 sB[2 * 64 * KC];    // double-buffered weight stage [n][k]

    const int t      = threadIdx.x;
    const int lane   = t & 31;
    const int wid    = t >> 5;
    const int lane16 = lane & 15;
    const int hf     = lane >> 4;
    const int wm     = wid >> 2;
    const int wn     = wid & 3;
    const int rowBase = blockIdx.x * MTILE;
    const uint32_t sB_lds = (uint32_t)(uintptr_t)(&sB[0]);  // LDS byte address

    // ---- stage x tile: fp32 -> bf16, once ----
    {
        const int rl = t >> 3;
        const int c8 = t & 7;
        const float* xr = x + (size_t)(rowBase + rl) * DK;
        #pragma unroll 4
        for (int i = 0; i < 32; ++i) {
            const int col = c8 * 4 + i * 32;
            v4f d = *(const v4f*)(xr + col);
            sA[rl * DK + col + 0] = (__bf16)d.x;
            sA[rl * DK + col + 1] = (__bf16)d.y;
            sA[rl * DK + col + 2] = (__bf16)d.z;
            sA[rl * DK + col + 3] = (__bf16)d.w;
        }
    }

    const float* Wmats[3]  = { Wq, Wk, Wv };
    const float* biases[3] = { bq, bk, bv };

    // ---- phase 1: QKV GEMM, N in 48 chunks of 64 ----
    for (int c = 0; c < 48; ++c) {
        const int mat = c >> 4;
        const int nc0 = (c & 15) * 64;
        v8f acc = {};

        const uint64_t saddr = USE_WT
            ? (uint64_t)(uintptr_t)(wt + (size_t)mat * 1024 * 1024) : 0;
        const float* Wp = Wmats[mat];

        // async stage of one 64n x 128k bf16 block into sB[buf]
        auto stage_async = [&](int buf, int kb0) {
            #pragma unroll
            for (int j = 0; j < 4; ++j) {
                const int id = t + j * 256;            // 0..1023 b128 chunks
                const int n  = id >> 4;                // 0..63
                const int ko = (id & 15) * 8;          // halves within row
                const uint32_t voff  = (uint32_t)((nc0 + n) * 2048 + (kb0 + ko) * 2);
                const uint32_t laddr = sB_lds + (uint32_t)(buf * 16384 + n * 256 + ko * 2);
                asm volatile("global_load_async_to_lds_b128 %0, %1, %2"
                             :: "v"(laddr), "v"(voff), "s"(saddr) : "memory");
            }
        };

        if (USE_WT) stage_async(0, 0);                 // prologue

        for (int i = 0; i < 8; ++i) {
            const int kb0 = i * KC;
            int bufbase;
            if (USE_WT) {
                asm volatile("s_wait_asynccnt 0x0" ::: "memory");
                __syncthreads();                       // stage i visible to all waves
                if (i < 7) stage_async((i + 1) & 1, kb0 + KC);  // overlap next copy
                bufbase = (i & 1) * 8192;
            } else {
                __syncthreads();
                #pragma unroll
                for (int j = 0; j < 8; ++j) {          // fallback: fp32 load + cvt
                    const int kl = j * 16 + (t >> 4);
                    const int nl = (t & 15) * 4;
                    v4f d = *(const v4f*)(Wp + (size_t)(kb0 + kl) * 1024 + nc0 + nl);
                    sB[(nl + 0) * KC + kl] = (__bf16)d.x;
                    sB[(nl + 1) * KC + kl] = (__bf16)d.y;
                    sB[(nl + 2) * KC + kl] = (__bf16)d.z;
                    sB[(nl + 3) * KC + kl] = (__bf16)d.w;
                }
                __syncthreads();
                bufbase = 0;
            }

            const __bf16* sBc = sB + bufbase;
            const int arow = (wm * 16 + lane16) * DK + kb0;
            const int bcol = (wn * 16 + lane16) * KC;
            #pragma unroll
            for (int kk = 0; kk < 4; ++kk) {
                const int k0 = kk * 32;
                V16 a;                                  // A: lane=row, 16 consec k {0-15}/{16-31}
                a.h[0] = lds_ld8(&sA[arow + k0 + hf * 16]);
                a.h[1] = lds_ld8(&sA[arow + k0 + hf * 16 + 8]);
                v16bf b = lds_ld16(&sBc[bcol + k0 + hf * 16]);  // B: lane=col, 16 consec k
                acc = __builtin_amdgcn_wmma_f32_16x16x32_bf16(
                          false, a.v, false, b, (short)0, acc, false, false);
            }
        }

        // epilogue: bias + bf16 store into QKV tile
        const int colm = nc0 + wn * 16 + lane16;
        const float bb = biases[mat][colm];
        const int col3 = mat * 1024 + colm;
        const int r0 = wm * 16 + hf * 8;
        #pragma unroll
        for (int v = 0; v < 8; ++v)
            sQKV[(r0 + v) * N3 + col3] = (__bf16)(acc[v] + bb);
    }
    __syncthreads();

    // ---- phase 2: per-position attention ----
    __bf16* aw = &sB[wid * 512];                        // private 16x32 attn tile per wave
    #pragma unroll
    for (int i = 0; i < 16; ++i)
        aw[lane * 16 + i] = (__bf16)0.0f;               // zero once; k=16..31 pad stays 0

    for (int pi = 0; pi < 4; ++pi) {
        const int p = wid * 4 + pi;
        const __bf16* qrow = &sQKV[p * N3];

        v8f s = {};
        #pragma unroll
        for (int d0 = 0; d0 < 64; d0 += 32) {           // scores = Q_ @ K_^T
            V16 a;
            const __bf16* qh = qrow + lane16 * 64 + d0;
            a.h[0] = lds_ld8(qh + hf * 16);
            a.h[1] = lds_ld8(qh + hf * 16 + 8);
            v16bf kb16 = lds_ld16(qrow + 1024 + lane16 * 64 + d0 + hf * 16);
            s = __builtin_amdgcn_wmma_f32_16x16x32_bf16(
                    false, a.v, false, kb16, (short)0, s, false, false);
        }

        float pr[8];                                    // softmax across 16-lane half
        #pragma unroll
        for (int v = 0; v < 8; ++v) {
            float xv = s[v];
            float m = xv;
            m = fmaxf(m, __shfl_xor(m, 1, 32));
            m = fmaxf(m, __shfl_xor(m, 2, 32));
            m = fmaxf(m, __shfl_xor(m, 4, 32));
            m = fmaxf(m, __shfl_xor(m, 8, 32));
            float e = __expf(xv - m);
            float su = e;
            su += __shfl_xor(su, 1, 32);
            su += __shfl_xor(su, 2, 32);
            su += __shfl_xor(su, 4, 32);
            su += __shfl_xor(su, 8, 32);
            pr[v] = e / su;
        }

        #pragma unroll
        for (int v = 0; v < 8; ++v)                     // C/D layout -> A layout via LDS
            aw[(v + hf * 8) * 32 + lane16] = (__bf16)pr[v];
        asm volatile("s_wait_dscnt 0x0" ::: "memory");
        V16 at;
        at.h[0] = lds_ld8(aw + lane16 * 32 + hf * 16);
        at.h[1] = lds_ld8(aw + lane16 * 32 + hf * 16 + 8);

        float* orow = out + (size_t)(rowBase + p) * 1024;
        #pragma unroll
        for (int dc = 0; dc < 4; ++dc) {                // out = probs @ V_
            v16bf bv16;
            #pragma unroll
            for (int e = 0; e < 16; ++e) {
                __bf16 val = qrow[2048 + e * 64 + dc * 16 + lane16];
                bv16[e] = hf ? (__bf16)0.0f : val;
            }
            v8f o = {};
            o = __builtin_amdgcn_wmma_f32_16x16x32_bf16(
                    false, at.v, false, bv16, (short)0, o, false, false);
            #pragma unroll
            for (int v = 0; v < 8; ++v)
                orow[(v + hf * 8) * 64 + dc * 16 + lane16] = o[v];
        }
    }
}

extern "C" void kernel_launch(void* const* d_in, const int* in_sizes, int n_in,
                              void* d_out, int out_size, void* d_ws, size_t ws_size,
                              hipStream_t stream) {
    const float* x  = (const float*)d_in[0];
    const float* Wq = (const float*)d_in[1];
    const float* bq = (const float*)d_in[2];
    const float* Wk = (const float*)d_in[3];
    const float* bk = (const float*)d_in[4];
    const float* Wv = (const float*)d_in[5];
    const float* bv = (const float*)d_in[6];
    float* out = (float*)d_out;

    const int M = 16 * 4096;
    dim3 grid(M / MTILE);
    dim3 block(BDIM);

    const size_t wt_bytes = (size_t)3 * 1024 * 1024 * 2;   // bf16 weights
    if (ws_size >= wt_bytes) {
        __bf16* wt = (__bf16*)d_ws;
        hipLaunchKernelGGL(weight_tr_cvt_kernel, dim3(768), dim3(256), 0, stream,
                           Wq, Wk, Wv, wt);
        hipLaunchKernelGGL((fused_mha_kernel<true>), grid, block, 0, stream,
                           x, Wq, bq, Wk, bk, Wv, bv, (const __bf16*)wt, out);
    } else {
        hipLaunchKernelGGL((fused_mha_kernel<false>), grid, block, 0, stream,
                           x, Wq, bq, Wk, bk, Wv, bv, (const __bf16*)nullptr, out);
    }
}
